// LSTM_3272765080013
// MI455X (gfx1250) — matrix operand order repaired and checked by reading, not compiled, not run on hardware
//
#include <hip/hip_runtime.h>

// ---------------------------------------------------------------------------
// 2-layer LSTM (B=256,T=256,I=64,H=512) + final linear, for gfx1250 (MI455X).
//
//  * Persistent scan kernel: 16 WGs (one per 16-batch-row tile), 16 waves;
//    each wave owns 32 h-columns (4 gates x 2 N-tiles). h double-buffered in
//    LDS (2 x 16x512 f16 = 32KB), one barrier per timestep.
//  * Input GEMM fused into every step -> no gx materialization.
//  * f16 WMMA (v_wmma_f32_16x16x32_f16, f32 accum); weights packed once into
//    B-fragment order, streamed from L2 each step.
//  * An opaque (asm-laundered) zero OFFSET is added to the weight pointers
//    inside the time loop: blocks LICM from hoisting the address-invariant
//    weight loads (which caused mass spills in rev 1) while preserving
//    pointer provenance so they stay global_load_b128 (rev 2's integer
//    pointer laundering degraded them to flat_load_b128, which ties up both
//    LOADcnt and DScnt and the LDS port).
// ---------------------------------------------------------------------------

typedef __attribute__((ext_vector_type(16))) _Float16 v16h;
typedef __attribute__((ext_vector_type(8)))  _Float16 v8h;
typedef __attribute__((ext_vector_type(8)))  float    v8f;

static constexpr int Bsz = 256;
static constexpr int Tsz = 256;
static constexpr int Isz = 64;
static constexpr int Hsz = 512;
static constexpr int NT  = 128;    // (4*H) / 16 N-tiles

union AFrag { v16h v; v8h h[2]; };
union BFrag { v16h v; v8h h[2]; };

// Opaque zero offset: defeats LICM on loads whose address is otherwise
// loop-invariant, without destroying the base pointer's address space.
__device__ __forceinline__ long launder_off() {
    long z = 0;
    asm volatile("" : "+s"(z));
    return z;
}

__device__ __forceinline__ float sigmoidf_(float x) {
    return 1.0f / (1.0f + __expf(-x));          // exp overflow -> 1/inf = 0, safe
}
__device__ __forceinline__ float tanhf_(float x) {
    float u = __expf(-2.0f * fabsf(x));          // u in (0,1], no overflow
    float t = 1.0f - 2.0f * u / (1.0f + u);
    return copysignf(t, x);
}

// ---- prep: fp32 -> fp16 elementwise convert (for x) ------------------------
__global__ void cvt_f16_kernel(const float* __restrict__ in,
                               _Float16* __restrict__ out, int n) {
    int i = blockIdx.x * blockDim.x + threadIdx.x;
    if (i < n) out[i] = (_Float16)in[i];
}

// ---- prep: pack W[4H,K] (row-major f32) into WMMA B-fragment order ---------
// B tile (kt,nt) is 32(K)x16(N): lane L holds column n = nt*16 + (L&15),
// k = kt*32 + ((L>>4)<<3) + {0..7, 16..23}, stored as 16 consecutive f16 per lane.
// Flat: pW[((nt*nKt + kt)*32 + lane)*16 + v]
__global__ void pack_w_kernel(const float* __restrict__ W,
                              _Float16* __restrict__ out, int K, int nKt) {
    int idx = blockIdx.x * blockDim.x + threadIdx.x; // (nt*nKt+kt)*32 + lane
    int total = NT * nKt * 32;
    if (idx >= total) return;
    int lane = idx & 31;
    int kt   = (idx >> 5) % nKt;
    int nt   = (idx >> 5) / nKt;
    int n     = nt * 16 + (lane & 15);
    int kbase = kt * 32 + ((lane >> 4) << 3);
    const float* wrow = W + (size_t)n * K + kbase;
    _Float16* o = out + (size_t)idx * 16;
#pragma unroll
    for (int v = 0; v < 16; ++v) o[v] = (_Float16)wrow[v + ((v >> 3) << 3)];
}

// ---- persistent LSTM layer scan -------------------------------------------
// grid = 16 (batch tiles of 16 rows), block = 512 (16 waves).
template <int NKT_IN, bool WRITE_SEQ, bool WRITE_LAST>
__launch_bounds__(512, 1)
__global__ void lstm_scan_kernel(
    const _Float16* __restrict__ Asrc,  // input sequence, f16
    long sAb, long sAt,                 // element strides: row(b), time(t)
    const _Float16* __restrict__ pWin,  // packed input weights  [NT][NKT_IN][32][16]
    const _Float16* __restrict__ pWhh,  // packed recur. weights [NT][16][32][16]
    const float* __restrict__ bih, const float* __restrict__ bhh,
    _Float16* __restrict__ hseq,        // [T][B][H] f16 (layer0 output) or unused
    float* __restrict__ hlast)          // [B][H] f32 (last step) or unused
{
    __shared__ _Float16 hbuf[2][16 * Hsz];      // double-buffered h, 32KB

    const int tid   = threadIdx.x;
    const int lane  = tid & 31;
    const int wave  = tid >> 5;
    const int btile = blockIdx.x;
    const int grp   = lane >> 4;                // 0/1 half-wave
    const int lm    = lane & 15;

    // h_0 = 0
    for (int i = tid; i < 16 * Hsz; i += 512) hbuf[0][i] = (_Float16)0.0f;
    __syncthreads();

    // per-lane bias (constant over time): q = gate*2 + j
    float bias[8];
#pragma unroll
    for (int g = 0; g < 4; ++g)
#pragma unroll
        for (int j = 0; j < 2; ++j) {
            int n = g * Hsz + wave * 32 + j * 16 + lm;
            bias[g * 2 + j] = bih[n] + bhh[n];
        }

    // c state in C-fragment layout, per j-tile
    float cst[2][8];
#pragma unroll
    for (int j = 0; j < 2; ++j)
#pragma unroll
        for (int r = 0; r < 8; ++r) cst[j][r] = 0.0f;

    for (int t = 0; t < Tsz; ++t) {
        const int cur = t & 1, nxt = cur ^ 1;

        v8f acc[8];
#pragma unroll
        for (int q = 0; q < 8; ++q) {
            float b = bias[q];
            v8f a;
#pragma unroll
            for (int r = 0; r < 8; ++r) a[r] = b;
            acc[q] = a;
        }

        // fresh (non-hoistable) weight pointers for this timestep; the
        // opaque zero offset keeps provenance -> global_load, not flat_load.
        const _Float16* pWinL = pWin + launder_off();
        const _Float16* pWhhL = pWhh + launder_off();

        // ---- phase 1: input contribution (A from global f16 sequence) ----
        {
            const _Float16* rowp =
                Asrc + (long)(btile * 16 + lm) * sAb + (long)t * sAt;
#pragma unroll
            for (int kt = 0; kt < NKT_IN; ++kt) {
                AFrag A;
                const _Float16* p = rowp + kt * 32 + grp * 8;
                A.h[0] = *(const v8h*)(p);
                A.h[1] = *(const v8h*)(p + 16);
#pragma unroll
                for (int q = 0; q < 8; ++q) {
                    const int g = q >> 1, j = q & 1;
                    const int nt = g * 32 + wave * 2 + j;
                    BFrag Bf;
                    const _Float16* wp =
                        pWinL + ((size_t)(nt * NKT_IN + kt) * 32 + lane) * 16;
                    Bf.h[0] = *(const v8h*)(wp);
                    Bf.h[1] = *(const v8h*)(wp + 8);
                    acc[q] = __builtin_amdgcn_wmma_f32_16x16x32_f16(
                        false, A.v, false, Bf.v, (short)0, acc[q], false, false);
                }
            }
        }

        // ---- phase 2: recurrent contribution (A from LDS h) --------------
        {
            const _Float16* rowp = &hbuf[cur][lm * Hsz];
#pragma unroll
            for (int kt = 0; kt < 16; ++kt) {
                if (kt + 1 < 16)  // warm next K-tile's weight stream in WGP$
                    __builtin_prefetch(
                        pWhhL + ((size_t)((wave * 2) * 16 + (kt + 1)) * 32 + lane) * 16,
                        0, 0);
                AFrag A;
                const _Float16* p = rowp + kt * 32 + grp * 8;
                A.h[0] = *(const v8h*)(p);
                A.h[1] = *(const v8h*)(p + 16);
#pragma unroll
                for (int q = 0; q < 8; ++q) {
                    const int g = q >> 1, j = q & 1;
                    const int nt = g * 32 + wave * 2 + j;
                    BFrag Bf;
                    const _Float16* wp =
                        pWhhL + ((size_t)(nt * 16 + kt) * 32 + lane) * 16;
                    Bf.h[0] = *(const v8h*)(wp);
                    Bf.h[1] = *(const v8h*)(wp + 8);
                    acc[q] = __builtin_amdgcn_wmma_f32_16x16x32_f16(
                        false, A.v, false, Bf.v, (short)0, acc[q], false, false);
                }
            }
        }

        // ---- gates + state update (i,f,g,o in PyTorch order) -------------
#pragma unroll
        for (int j = 0; j < 2; ++j) {
            const int ncol = wave * 32 + j * 16 + lm;
#pragma unroll
            for (int r = 0; r < 8; ++r) {
                float ig = sigmoidf_(acc[0 + j][r]);
                float fg = sigmoidf_(acc[2 + j][r]);
                float gg = tanhf_   (acc[4 + j][r]);
                float og = sigmoidf_(acc[6 + j][r]);
                float c  = fg * cst[j][r] + ig * gg;
                cst[j][r] = c;
                float h  = og * tanhf_(c);
                const int m = (grp << 3) + r;
                hbuf[nxt][m * Hsz + ncol] = (_Float16)h;
                if (WRITE_SEQ)
                    hseq[((size_t)t * Bsz + (btile * 16 + m)) * Hsz + ncol] =
                        (_Float16)h;
                if (WRITE_LAST)
                    if (t == Tsz - 1)
                        hlast[(size_t)(btile * 16 + m) * Hsz + ncol] = h;
            }
        }
        __syncthreads();   // publish hbuf[nxt] before next step reads it
    }
}

// ---- final linear: out[b] = dot(h2[b,:], Wfc[0,:]) + bfc[0] ----------------
__global__ void fc_kernel(const float* __restrict__ h2,
                          const float* __restrict__ Wfc,
                          const float* __restrict__ bfc,
                          float* __restrict__ out) {
    int b = blockIdx.x;
    int lane = threadIdx.x;  // 32 threads (one wave32)
    float s = 0.0f;
    for (int i = lane; i < Hsz; i += 32)
        s += h2[(size_t)b * Hsz + i] * Wfc[i];
#pragma unroll
    for (int off = 16; off > 0; off >>= 1) s += __shfl_down(s, off, 32);
    if (lane == 0) out[b] = s + bfc[0];
}

// ---------------------------------------------------------------------------
extern "C" void kernel_launch(void* const* d_in, const int* in_sizes, int n_in,
                              void* d_out, int out_size, void* d_ws, size_t ws_size,
                              hipStream_t stream) {
    const float* x    = (const float*)d_in[0];
    const float* Wih0 = (const float*)d_in[1];
    const float* Whh0 = (const float*)d_in[2];
    const float* bih0 = (const float*)d_in[3];
    const float* bhh0 = (const float*)d_in[4];
    const float* Wih1 = (const float*)d_in[5];
    const float* Whh1 = (const float*)d_in[6];
    const float* bih1 = (const float*)d_in[7];
    const float* bhh1 = (const float*)d_in[8];
    const float* Wfc  = (const float*)d_in[9];
    const float* bfc  = (const float*)d_in[10];
    float* out = (float*)d_out;

    char* ws = (char*)d_ws;
    size_t off = 0;
    auto take = [&](size_t bytes) -> char* {
        char* p = ws + off;
        off += (bytes + 255) & ~(size_t)255;
        return p;
    };
    _Float16* xf16  = (_Float16*)take((size_t)Bsz * Tsz * Isz * 2);  //  8 MB
    _Float16* pWih0 = (_Float16*)take((size_t)NT * 2  * 32 * 16 * 2);
    _Float16* pWhh0 = (_Float16*)take((size_t)NT * 16 * 32 * 16 * 2);
    _Float16* pWih1 = (_Float16*)take((size_t)NT * 16 * 32 * 16 * 2);
    _Float16* pWhh1 = (_Float16*)take((size_t)NT * 16 * 32 * 16 * 2);
    _Float16* h1    = (_Float16*)take((size_t)Tsz * Bsz * Hsz * 2);  // 64 MB
    float*    h2    = (float*)   take((size_t)Bsz * Hsz * 4);
    (void)ws_size; (void)in_sizes; (void)n_in; (void)out_size;

    // prep: convert x to f16, pack all weight matrices into B-fragment order
    const int nx = Bsz * Tsz * Isz;
    cvt_f16_kernel<<<(nx + 255) / 256, 256, 0, stream>>>(x, xf16, nx);
    pack_w_kernel<<<(NT * 2  * 32) / 256, 256, 0, stream>>>(Wih0, pWih0, Isz, 2);
    pack_w_kernel<<<(NT * 16 * 32) / 256, 256, 0, stream>>>(Whh0, pWhh0, Hsz, 16);
    pack_w_kernel<<<(NT * 16 * 32) / 256, 256, 0, stream>>>(Wih1, pWih1, Hsz, 16);
    pack_w_kernel<<<(NT * 16 * 32) / 256, 256, 0, stream>>>(Whh1, pWhh1, Hsz, 16);

    // layer 0: A = xf16[b,t,:], addr = (b*T + t)*I  -> sAb = T*I, sAt = I
    lstm_scan_kernel<2, true, false><<<Bsz / 16, 512, 0, stream>>>(
        xf16, (long)(Tsz * Isz), (long)Isz, pWih0, pWhh0, bih0, bhh0,
        h1, (float*)nullptr);

    // layer 1: A = h1[t,b,:], addr = (t*B + b)*H -> sAb = H, sAt = B*H
    lstm_scan_kernel<16, false, true><<<Bsz / 16, 512, 0, stream>>>(
        h1, (long)Hsz, (long)(Bsz * Hsz), pWih1, pWhh1, bih1, bhh1,
        (_Float16*)nullptr, h2);

    // final linear on last timestep
    fc_kernel<<<Bsz, 32, 0, stream>>>(h2, Wfc, bfc, out);
}
